// TrajectoryPredictorRNN_18803366822181
// MI455X (gfx1250) — matrix operand order repaired and checked by reading, not compiled, not run on hardware
//
#include <hip/hip_runtime.h>

#define B_TOTAL 8192
#define T_HIST  11
#define T_PRED  80
#define IN_DIM  2
#define H       64
#define HP      72                 // padded LDS row stride (bf16 elems); 144B rows keep 16B alignment
#define WAVES   8
#define GROUPS  (WAVES / 2)        // 2 waves cooperate on one 16-row batch tile
#define ROWS_PER_GROUP 16
#define ROWS_PER_BLOCK (GROUPS * ROWS_PER_GROUP)

typedef __attribute__((ext_vector_type(16))) __bf16 v16bf;
typedef __attribute__((ext_vector_type(8)))  float  v8f;

union BFrag {
  v16bf v;
  unsigned short s[16];
  int4 q[2];
};

__device__ __forceinline__ unsigned short f2bf(float f) {
  unsigned int u = __float_as_uint(f);
  u += 0x7FFFu + ((u >> 16) & 1u);          // round-to-nearest-even
  return (unsigned short)(u >> 16);
}

__device__ __forceinline__ v8f wmma_bf16(v16bf a, v16bf b, v8f c) {
  // D = A(16x32 bf16) * B(32x16 bf16) + C(16x16 f32)
  return __builtin_amdgcn_wmma_f32_16x16x32_bf16(false, a, false, b, (short)0, c, false, false);
}

// B fragment (32x16 bf16) of W^T from row-major fp32 W[rows][cols].
// B(k,n) = W[n][k]. Lane (n = lane&15) holds 16 contiguous K starting at kt*32 + (lane>=16)*16.
__device__ __forceinline__ v16bf ldBfragW(const float* __restrict__ W, int cols,
                                          int nt, int kt, int lane) {
  int n  = nt * 16 + (lane & 15);
  int k0 = kt * 32 + ((lane >> 4) << 4);
  const float* p = W + n * cols + k0;
  BFrag r;
#pragma unroll
  for (int i = 0; i < 16; ++i) r.s[i] = f2bf(p[i]);
  return r.v;
}

// fc_W is (2 x 64); columns n>=2 of the B fragment are zero.
__device__ __forceinline__ v16bf ldFcFrag(const float* __restrict__ W, int kt, int lane) {
  int n  = lane & 15;
  int k0 = kt * 32 + ((lane >> 4) << 4);
  BFrag r;
#pragma unroll
  for (int i = 0; i < 16; ++i) r.s[i] = 0;
  if (n < 2) {
    const float* p = W + n * H + k0;
#pragma unroll
    for (int i = 0; i < 16; ++i) r.s[i] = f2bf(p[i]);
  }
  return r.v;
}

// A fragment (16x32 bf16) from LDS row-major h[16][HP].
// elems 0..15 : K = kt*32 + hi*16 + i
__device__ __forceinline__ v16bf ldAfrag(const unsigned short* h, int kt, int lane) {
  int m  = lane & 15;
  int hi = lane >> 4;
  const unsigned short* p = h + m * HP + kt * 32 + hi * 16;
  BFrag r;
  r.q[0] = *(const int4*)(p);
  r.q[1] = *(const int4*)(p + 8);
  return r.v;
}

// Store a 16x16 f32 accumulator tile (lane->N, vgpr->M) as bf16 into LDS row-major.
__device__ __forceinline__ void stAcc(unsigned short* h, v8f a, int nt, int lane) {
  int n  = nt * 16 + (lane & 15);
  int mb = (lane >> 4) << 3;
#pragma unroll
  for (int r = 0; r < 8; ++r) h[(mb + r) * HP + n] = f2bf(a[r]);
}

__device__ __forceinline__ v8f tanh8(v8f a) {
#pragma unroll
  for (int r = 0; r < 8; ++r) {
    float xv = a[r];
    float e  = __expf(-2.0f * fabsf(xv));                   // e in (0,1], no overflow
    float t  = (1.0f - e) * __builtin_amdgcn_rcpf(1.0f + e);
    a[r] = copysignf(t, xv);
  }
  return a;
}

__launch_bounds__(WAVES * 32)
__global__ void rnn_fused(const float* __restrict__ x,
    const float* __restrict__ eW0, const float* __restrict__ eU0,
    const float* __restrict__ eb0i, const float* __restrict__ eb0h,
    const float* __restrict__ eW1, const float* __restrict__ eU1,
    const float* __restrict__ eb1i, const float* __restrict__ eb1h,
    const float* __restrict__ dW0, const float* __restrict__ dU0,
    const float* __restrict__ db0i, const float* __restrict__ db0h,
    const float* __restrict__ dW1, const float* __restrict__ dU1,
    const float* __restrict__ db1i, const float* __restrict__ db1h,
    const float* __restrict__ fcW, const float* __restrict__ fcb,
    float* __restrict__ out)
{
  __shared__ __align__(16) unsigned short sH1[GROUPS][ROWS_PER_GROUP * HP];
  __shared__ __align__(16) unsigned short sH2[GROUPS][ROWS_PER_GROUP * HP];
  __shared__ float sInp[GROUPS][34];   // [m*2+j] for m<16,j<2 ; slot 32 = divergence dump

  const int tid   = threadIdx.x;
  const int wave  = tid >> 5;
  const int group = wave >> 1;         // pair of waves per 16-row tile
  const int p     = wave & 1;          // parity: which half of N this wave owns
  const int lane  = tid & 31;
  const int lo    = lane & 15;
  const int hi    = lane >> 4;
  const int rowBase = blockIdx.x * ROWS_PER_BLOCK + group * ROWS_PER_GROUP;

  unsigned short* h1w = sH1[group];
  unsigned short* h2w = sH2[group];
  float* inpw = sInp[group];

  // h0 = 0 : the two waves of the group split the clear
  {
    int tg = tid & 63;
    for (int i = tg; i < ROWS_PER_GROUP * HP; i += 64) { h1w[i] = 0; h2w[i] = 0; }
  }

  // ============ encoder phase: this wave's half of each weight matrix in VGPRs ============
  v16bf eU0f[2][2], eW1f[2][2], eU1f[2][2];
#pragma unroll
  for (int kt = 0; kt < 2; ++kt)
#pragma unroll
    for (int nt = 0; nt < 2; ++nt) {
      int ntg = p * 2 + nt;
      eU0f[kt][nt] = ldBfragW(eU0, H, ntg, kt, lane);
      eW1f[kt][nt] = ldBfragW(eW1, H, ntg, kt, lane);
      eU1f[kt][nt] = ldBfragW(eU1, H, ntg, kt, lane);
    }
  float be0[2], be1[2], ew00[2], ew01[2];
#pragma unroll
  for (int nt = 0; nt < 2; ++nt) {
    int n = (p * 2 + nt) * 16 + lo;
    be0[nt]  = eb0i[n] + eb0h[n];
    be1[nt]  = eb1i[n] + eb1h[n];
    ew00[nt] = eW0[n * IN_DIM + 0];
    ew01[nt] = eW0[n * IN_DIM + 1];
  }
  __syncthreads();  // h zero-init visible to both waves of the group

  for (int t = 0; t < T_HIST; ++t) {
    v16bf a1k0 = ldAfrag(h1w, 0, lane), a1k1 = ldAfrag(h1w, 1, lane);
    v16bf a2k0 = ldAfrag(h2w, 0, lane), a2k1 = ldAfrag(h2w, 1, lane);
    float i0[8], i1[8];
#pragma unroll
    for (int r = 0; r < 8; ++r) {
      const float* xp = x + ((rowBase + hi * 8 + r) * T_HIST + t) * IN_DIM;
      i0[r] = xp[0]; i1[r] = xp[1];
    }
    // layer 0: h1 = tanh(x @ W0^T + b + h1 @ U0^T)   (this wave: 2 N-tiles)
#pragma unroll
    for (int nt = 0; nt < 2; ++nt) {
      v8f acc;
#pragma unroll
      for (int r = 0; r < 8; ++r) acc[r] = be0[nt] + i0[r] * ew00[nt] + i1[r] * ew01[nt];
      acc = wmma_bf16(a1k0, eU0f[0][nt], acc);
      acc = wmma_bf16(a1k1, eU0f[1][nt], acc);
      acc = tanh8(acc);
      stAcc(h1w, acc, p * 2 + nt, lane);
    }
    __syncthreads();  // full h1_new assembled from both waves
    // layer 1: h2 = tanh(h1_new @ W1^T + b + h2 @ U1^T)
    v16bf n1k0 = ldAfrag(h1w, 0, lane), n1k1 = ldAfrag(h1w, 1, lane);
#pragma unroll
    for (int nt = 0; nt < 2; ++nt) {
      v8f acc;
#pragma unroll
      for (int r = 0; r < 8; ++r) acc[r] = be1[nt];
      acc = wmma_bf16(n1k0, eW1f[0][nt], acc);
      acc = wmma_bf16(n1k1, eW1f[1][nt], acc);
      acc = wmma_bf16(a2k0, eU1f[0][nt], acc);
      acc = wmma_bf16(a2k1, eU1f[1][nt], acc);
      acc = tanh8(acc);
      stAcc(h2w, acc, p * 2 + nt, lane);
    }
    __syncthreads();  // full h2_new before next step reads it
  }

  // ================= decoder phase =================
  v16bf dU0f[2][2], dW1f[2][2], dU1f[2][2], fcf[2];
#pragma unroll
  for (int kt = 0; kt < 2; ++kt) {
#pragma unroll
    for (int nt = 0; nt < 2; ++nt) {
      int ntg = p * 2 + nt;
      dU0f[kt][nt] = ldBfragW(dU0, H, ntg, kt, lane);
      dW1f[kt][nt] = ldBfragW(dW1, H, ntg, kt, lane);
      dU1f[kt][nt] = ldBfragW(dU1, H, ntg, kt, lane);
    }
    fcf[kt] = ldFcFrag(fcW, kt, lane);   // loaded by both waves: keeps WMMA non-divergent
  }
  float bd0[2], bd1[2], dw00[2], dw01[2];
#pragma unroll
  for (int nt = 0; nt < 2; ++nt) {
    int n = (p * 2 + nt) * 16 + lo;
    bd0[nt]  = db0i[n] + db0h[n];
    bd1[nt]  = db1i[n] + db1h[n];
    dw00[nt] = dW0[n * IN_DIM + 0];
    dw01[nt] = dW0[n * IN_DIM + 1];
  }
  float fcbl = (lo < 2) ? fcb[lo] : 0.0f;

  // seed decoder input with x[:, T_HIST-1, :]  (wave p==0 of each group)
  if (p == 0) {
    int m = lane >> 1, j = lane & 1;
    inpw[m * 2 + j] = x[(rowBase + m) * T_HIST * IN_DIM + (T_HIST - 1) * IN_DIM + j];
  }
  __syncthreads();

  for (int t = 0; t < T_PRED; ++t) {
    v16bf a1k0 = ldAfrag(h1w, 0, lane), a1k1 = ldAfrag(h1w, 1, lane);
    v16bf a2k0 = ldAfrag(h2w, 0, lane), a2k1 = ldAfrag(h2w, 1, lane);
    float i0[8], i1[8];
#pragma unroll
    for (int r = 0; r < 8; ++r) {
      int m = hi * 8 + r;
      i0[r] = inpw[m * 2 + 0];
      i1[r] = inpw[m * 2 + 1];
    }
    // layer 0
#pragma unroll
    for (int nt = 0; nt < 2; ++nt) {
      v8f acc;
#pragma unroll
      for (int r = 0; r < 8; ++r) acc[r] = bd0[nt] + i0[r] * dw00[nt] + i1[r] * dw01[nt];
      acc = wmma_bf16(a1k0, dU0f[0][nt], acc);
      acc = wmma_bf16(a1k1, dU0f[1][nt], acc);
      acc = tanh8(acc);
      stAcc(h1w, acc, p * 2 + nt, lane);
    }
    __syncthreads();
    // layer 1
    v16bf n1k0 = ldAfrag(h1w, 0, lane), n1k1 = ldAfrag(h1w, 1, lane);
#pragma unroll
    for (int nt = 0; nt < 2; ++nt) {
      v8f acc;
#pragma unroll
      for (int r = 0; r < 8; ++r) acc[r] = bd1[nt];
      acc = wmma_bf16(n1k0, dW1f[0][nt], acc);
      acc = wmma_bf16(n1k1, dW1f[1][nt], acc);
      acc = wmma_bf16(a2k0, dU1f[0][nt], acc);
      acc = wmma_bf16(a2k1, dU1f[1][nt], acc);
      acc = tanh8(acc);
      stAcc(h2w, acc, p * 2 + nt, lane);
    }
    __syncthreads();
    // readout: pred = h2_new @ fc_W^T + fc_b (columns n<2 valid).
    // Executed redundantly by BOTH waves so EXEC stays all-1s around WMMA;
    // only wave p==0 commits results.
    v16bf n2k0 = ldAfrag(h2w, 0, lane), n2k1 = ldAfrag(h2w, 1, lane);
    v8f accp;
#pragma unroll
    for (int r = 0; r < 8; ++r) accp[r] = fcbl;
    accp = wmma_bf16(n2k0, fcf[0], accp);
    accp = wmma_bf16(n2k1, fcf[1], accp);
    // unconditional LDS write (non-owning lanes -> dump slot) consumes accp outside divergence
#pragma unroll
    for (int r = 0; r < 8; ++r) {
      int m   = hi * 8 + r;
      int idx = (p == 0 && lo < 2) ? (m * 2 + lo) : 32;
      inpw[idx] = accp[r];
    }
    if (p == 0 && lo < 2) {
#pragma unroll
      for (int r = 0; r < 8; ++r) {
        int m = hi * 8 + r;
        out[(rowBase + m) * (T_PRED * IN_DIM) + t * IN_DIM + lo] = accp[r];
      }
    }
    __syncthreads();  // inpw ready for next step's layer-0 input
  }
}

extern "C" void kernel_launch(void* const* d_in, const int* in_sizes, int n_in,
                              void* d_out, int out_size, void* d_ws, size_t ws_size,
                              hipStream_t stream) {
  (void)in_sizes; (void)n_in; (void)d_ws; (void)ws_size; (void)out_size;
  const float* x    = (const float*)d_in[0];
  const float* eW0  = (const float*)d_in[1];
  const float* eU0  = (const float*)d_in[2];
  const float* eb0i = (const float*)d_in[3];
  const float* eb0h = (const float*)d_in[4];
  const float* eW1  = (const float*)d_in[5];
  const float* eU1  = (const float*)d_in[6];
  const float* eb1i = (const float*)d_in[7];
  const float* eb1h = (const float*)d_in[8];
  const float* dW0  = (const float*)d_in[9];
  const float* dU0  = (const float*)d_in[10];
  const float* db0i = (const float*)d_in[11];
  const float* db0h = (const float*)d_in[12];
  const float* dW1  = (const float*)d_in[13];
  const float* dU1  = (const float*)d_in[14];
  const float* db1i = (const float*)d_in[15];
  const float* db1h = (const float*)d_in[16];
  const float* fcW  = (const float*)d_in[17];
  const float* fcb  = (const float*)d_in[18];
  float* out = (float*)d_out;

  dim3 grid(B_TOTAL / ROWS_PER_BLOCK);
  dim3 block(WAVES * 32);
  hipLaunchKernelGGL(rnn_fused, grid, block, 0, stream,
                     x, eW0, eU0, eb0i, eb0h, eW1, eU1, eb1i, eb1h,
                     dW0, dU0, db0i, db0h, dW1, dU1, db1i, db1h,
                     fcW, fcb, out);
}
